// HGNN_23089744183780
// MI455X (gfx1250) — matrix operand order repaired and checked by reading, not compiled, not run on hardware
//
#include <hip/hip_runtime.h>
#include <hip/hip_bf16.h>
#include <string.h>

// ---------------------------------------------------------------------------
// HGNN forward for MI455X (gfx1250).
// - bf16 WMMA NT-GEMMs (C = A * Bt^T, both row-major, K contiguous)
// - gfx1250 async global->LDS (ASYNCcnt) with double-buffered LDS tiles
// - HHt never materialized: HHt@Xh = H~ @ (H~^T @ Xh)
// - block tile 64x128 (4 waves), wave tile 32x64 = 2x4 v_wmma_f32_16x16x32_bf16
// ---------------------------------------------------------------------------

#define HV   8192   // V nodes
#define HE   4096   // E hyperedges
#define DIN  512
#define DH   256

typedef __attribute__((ext_vector_type(16))) __bf16 v16bf;
typedef __attribute__((ext_vector_type(8)))  float  v8f;

union FragBF {
    v16bf v;
    uint4 u[2];
};

// 32-bit LDS byte offset of a generic pointer into shared memory
__device__ __forceinline__ unsigned lds_off(const void* p)
{
    return (unsigned)(size_t)(const __attribute__((address_space(3))) void*)p;
}

// one per-lane 16B async DMA chunk: global -> LDS, tracked by ASYNCcnt
__device__ __forceinline__ void async_b128(unsigned lds_byte_off, const void* gptr)
{
    asm volatile("global_load_async_to_lds_b128 %0, %1, off"
                 :: "v"(lds_byte_off), "v"(gptr) : "memory");
}
__device__ __forceinline__ void wait_async_6()
{
    asm volatile("s_wait_asynccnt 0x6" ::: "memory");
}
__device__ __forceinline__ void wait_async_0()
{
    asm volatile("s_wait_asynccnt 0x0" ::: "memory");
}

// ----------------------------- degree kernels ------------------------------

__global__ __launch_bounds__(256)
void degv_kernel(const float* __restrict__ H, float* __restrict__ dvi,
                 float* __restrict__ invdvi, int En)
{
    __shared__ float red[256];
    const int v = blockIdx.x;
    const int t = threadIdx.x;
    float s = 0.f;
    for (int e = t; e < En; e += 256) s += H[(size_t)v * En + e];
    red[t] = s;
    __syncthreads();
    for (int st = 128; st > 0; st >>= 1) {
        if (t < st) red[t] += red[t + st];
        __syncthreads();
    }
    if (t == 0) {
        float d = red[0];
        dvi[v]    = (d > 0.f) ? rsqrtf(d) : 0.f;
        invdvi[v] = (d > 0.f) ? sqrtf(d)  : 0.f;   // 1/dvi, guarded
    }
}

__global__ __launch_bounds__(256)
void dege_kernel(const float* __restrict__ H, float* __restrict__ dei, int Vn, int En)
{
    const int e = blockIdx.x * blockDim.x + threadIdx.x;
    if (e >= En) return;
    float s = 0.f;
    for (int v = 0; v < Vn; ++v) s += H[(size_t)v * En + e];
    dei[e] = (s > 0.f) ? rsqrtf(s) : 0.f;   // also the pooling row scale
}

// ------------------- build normalized H (bf16) + transpose -----------------

__global__ __launch_bounds__(256)
void build_H_kernel(const float* __restrict__ H,
                    const float* __restrict__ dvi,
                    const float* __restrict__ dei,
                    __bf16* __restrict__ Hb,    // (V,E) row-major
                    __bf16* __restrict__ HTb,   // (E,V) row-major
                    int Vn, int En)
{
    __shared__ __bf16 tile[32][33];
    const int v0 = blockIdx.y * 32;
    const int e0 = blockIdx.x * 32;
    const int tx = threadIdx.x & 31;
    const int ty = threadIdx.x >> 5;     // 0..7
    #pragma unroll
    for (int i = 0; i < 32; i += 8) {
        const int v = v0 + ty + i;
        const int e = e0 + tx;
        float h = H[(size_t)v * En + e] * dvi[v] * dei[e];
        __bf16 hb = (__bf16)h;
        Hb[(size_t)v * En + e] = hb;
        tile[ty + i][tx] = hb;
    }
    __syncthreads();
    #pragma unroll
    for (int i = 0; i < 32; i += 8)
        HTb[(size_t)(e0 + ty + i) * Vn + v0 + tx] = tile[tx][ty + i];
}

// ---------------------- small converts / transposes ------------------------

// src (R,C) f32 -> dst (C,R) bf16, optional per-src-row scale
__global__ __launch_bounds__(256)
void transpose_f32_to_bf16(const float* __restrict__ src, __bf16* __restrict__ dst,
                           const float* __restrict__ srcRowScale, int R, int C)
{
    __shared__ float tile[32][33];
    const int r0 = blockIdx.y * 32;
    const int c0 = blockIdx.x * 32;
    const int tx = threadIdx.x & 31;
    const int ty = threadIdx.x >> 5;
    #pragma unroll
    for (int i = 0; i < 32; i += 8) {
        const int r = r0 + ty + i;
        float v = src[(size_t)r * C + c0 + tx];
        if (srcRowScale) v *= srcRowScale[r];
        tile[ty + i][tx] = v;
    }
    __syncthreads();
    #pragma unroll
    for (int i = 0; i < 32; i += 8)
        dst[(size_t)(c0 + ty + i) * R + r0 + tx] = (__bf16)tile[tx][ty + i];
}

__global__ __launch_bounds__(256)
void convert_bf16_kernel(const float* __restrict__ src, __bf16* __restrict__ dst,
                         size_t n)
{
    size_t i = (size_t)blockIdx.x * blockDim.x + threadIdx.x;
    if (i < n) dst[i] = (__bf16)src[i];
}

// ------------------------------ WMMA NT-GEMM -------------------------------
// C[M,N] = A[M,K] * Bt[N,K]^T, A/Bt bf16 row-major, f32 accumulate.
// Block tile 64x128, K-step 32, 4 waves (128 threads) in a 2x2 grid,
// wave tile 32x64 = 2x4 v_wmma_f32_16x16x32_bf16 per K-step.
// Double-buffered LDS fed by global_load_async_to_lds_b128 (6 DMAs / wave /
// tile). M multiple of 64, N multiple of 128, K multiple of 32.

#define BM 64
#define BN 128
#define BK 32
#define TS 40   // padded bf16 row stride in LDS (conflict-free b128 frag reads)

__global__ __launch_bounds__(128)
void gemm_bf16_wmma(const __bf16* __restrict__ A, const __bf16* __restrict__ Bt,
                    float* __restrict__ outF, __bf16* __restrict__ outB,
                    const float* __restrict__ rowScale,
                    int M, int N, int K, int relu)
{
    __shared__ __align__(16) __bf16 As0[BM * TS], As1[BM * TS];   //  5 KB x2
    __shared__ __align__(16) __bf16 Bs0[BN * TS], Bs1[BN * TS];   // 10 KB x2

    const int tid  = threadIdx.x;     // 0..127
    const int wave = tid >> 5;        // 0..3
    const int lane = tid & 31;
    const int wr   = wave >> 1;       // wave row 0..1 (32 rows each)
    const int wc   = wave & 1;        // wave col 0..1 (64 cols each)
    const int row0 = blockIdx.y * BM;
    const int col0 = blockIdx.x * BN;
    const int half = lane >> 4;       // 0/1
    const int l15  = lane & 15;

    // per-thread DMA chunk coordinates (16B chunks, 4 per row of 32 bf16)
    const int r0c = tid >> 2;          // 0..31
    const int ccc = (tid & 3) * 8;     // 0/8/16/24

    v8f acc[2][4] = {};

    const int ntiles = K / BK;

    // ---- issue one K-tile (A: 2 chunks, Bt: 4 chunks per thread) ----
    auto issue_tile = [&](int k0, __bf16* As, __bf16* Bs) {
        #pragma unroll
        for (int i = 0; i < 2; ++i) {           // A rows r0c, r0c+32
            const int r = r0c + i * 32;
            async_b128(lds_off(As + r * TS + ccc),
                       A + (size_t)(row0 + r) * K + k0 + ccc);
        }
        #pragma unroll
        for (int i = 0; i < 4; ++i) {           // Bt rows r0c + {0,32,64,96}
            const int r = r0c + i * 32;
            async_b128(lds_off(Bs + r * TS + ccc),
                       Bt + (size_t)(col0 + r) * K + k0 + ccc);
        }
    };

    // ---- consume one K-tile: 2 A-frags, 4 B-frags, 8 WMMAs ----
    auto compute_tile = [&](const __bf16* As, const __bf16* Bs) {
        FragBF a[2], b[4];
        #pragma unroll
        for (int mt = 0; mt < 2; ++mt) {
            const __bf16* ap = As + (wr * 32 + mt * 16 + l15) * TS;
            a[mt].u[0] = *(const uint4*)(ap + half * 8);       // K = h*8 .. +7
            a[mt].u[1] = *(const uint4*)(ap + 16 + half * 8);  // K = 16+h*8 ..
        }
        #pragma unroll
        for (int nt = 0; nt < 4; ++nt) {
            const __bf16* bp = Bs + (wc * 64 + nt * 16 + l15) * TS;
            b[nt].u[0] = *(const uint4*)(bp + half * 8);       // K = h*8 .. +7
            b[nt].u[1] = *(const uint4*)(bp + 16 + half * 8);  // K = 16+h*8 ..
        }
        #pragma unroll
        for (int mt = 0; mt < 2; ++mt)
            #pragma unroll
            for (int nt = 0; nt < 4; ++nt)
                acc[mt][nt] = __builtin_amdgcn_wmma_f32_16x16x32_bf16(
                    false, a[mt].v, false, b[nt].v, (short)0, acc[mt][nt],
                    false, false);
    };

    issue_tile(0, As0, Bs0);
    // branch-free steady state; tail peeled
    for (int kt = 0; kt < ntiles - 1; ++kt) {
        __bf16* Asc = (kt & 1) ? As1 : As0;
        __bf16* Bsc = (kt & 1) ? Bs1 : Bs0;
        issue_tile((kt + 1) * BK, (kt & 1) ? As0 : As1, (kt & 1) ? Bs0 : Bs1);
        wait_async_6();               // this tile's 6 DMAs retired (in-order)
        __syncthreads();
        compute_tile(Asc, Bsc);
        __syncthreads();              // all waves done before buffer refill
    }
    wait_async_0();
    __syncthreads();
    compute_tile(((ntiles - 1) & 1) ? As1 : As0,
                 ((ntiles - 1) & 1) ? Bs1 : Bs0);

    // ---- store: C/D layout -> lane l holds (m = r + 8*(l>>4), n = l&15) ----
    #pragma unroll
    for (int mt = 0; mt < 2; ++mt) {
        #pragma unroll
        for (int nt = 0; nt < 4; ++nt) {
            #pragma unroll
            for (int r = 0; r < 8; ++r) {
                const int m = row0 + wr * 32 + mt * 16 + half * 8 + r;
                const int n = col0 + wc * 64 + nt * 16 + l15;
                float vv = acc[mt][nt][r];
                if (relu) vv = vv > 0.f ? vv : 0.f;
                if (rowScale) vv *= rowScale[m];
                if (outF) outF[(size_t)m * N + n] = vv;
                if (outB) outB[(size_t)m * N + n] = (__bf16)vv;
            }
        }
    }
}

// ------------------- residual + LayerNorm (row = 256) ----------------------

__global__ __launch_bounds__(256)
void lnres_kernel(float* __restrict__ Xh, const float* __restrict__ T,
                  const float* __restrict__ g, const float* __restrict__ b)
{
    __shared__ float red[256];
    const int v = blockIdx.x;
    const int t = threadIdx.x;
    float x = Xh[(size_t)v * DH + t] + T[(size_t)v * DH + t];
    red[t] = x;
    __syncthreads();
    for (int s = 128; s > 0; s >>= 1) {
        if (t < s) red[t] += red[t + s];
        __syncthreads();
    }
    const float m = red[0] * (1.f / DH);
    __syncthreads();
    const float d = x - m;
    red[t] = d * d;
    __syncthreads();
    for (int s = 128; s > 0; s >>= 1) {
        if (t < s) red[t] += red[t + s];
        __syncthreads();
    }
    const float var = red[0] * (1.f / DH);
    Xh[(size_t)v * DH + t] = d * rsqrtf(var + 1e-5f) * g[t] + b[t];
}

// --------------------- classifier + softmax (2 classes) --------------------

__global__ __launch_bounds__(256)
void cls_kernel(const float* __restrict__ zE, const float* __restrict__ Wc,
                const float* __restrict__ bc, float* __restrict__ out)
{
    __shared__ float r0[256], r1[256];
    const int e = blockIdx.x;
    const int t = threadIdx.x;
    const float z = zE[(size_t)e * DH + t];
    r0[t] = z * Wc[t];
    r1[t] = z * Wc[DH + t];
    __syncthreads();
    for (int s = 128; s > 0; s >>= 1) {
        if (t < s) { r0[t] += r0[t + s]; r1[t] += r1[t + s]; }
        __syncthreads();
    }
    if (t == 0) {
        const float l0 = r0[0] + bc[0];
        const float l1 = r1[0] + bc[1];
        const float mx = fmaxf(l0, l1);
        const float e0 = __expf(l0 - mx);
        const float e1 = __expf(l1 - mx);
        const float inv = 1.f / (e0 + e1);
        out[(size_t)e * 2 + 0] = e0 * inv;
        out[(size_t)e * 2 + 1] = e1 * inv;
    }
}

// ---------------------------------------------------------------------------

extern "C" void kernel_launch(void* const* d_in, const int* in_sizes, int n_in,
                              void* d_out, int out_size, void* d_ws, size_t ws_size,
                              hipStream_t stream)
{
    const float* X  = (const float*)d_in[0];   // (V,DIN)
    const float* H  = (const float*)d_in[1];   // (V,E)
    const float* Wp = (const float*)d_in[2];   // (DH,DIN)
    const float* W0 = (const float*)d_in[3];   // (DH,DH)
    const float* W1 = (const float*)d_in[4];   // (DH,DH)
    const float* g0 = (const float*)d_in[5];
    const float* b0 = (const float*)d_in[6];
    const float* g1 = (const float*)d_in[7];
    const float* b1 = (const float*)d_in[8];
    const float* Wc = (const float*)d_in[9];   // (2,DH)
    const float* bc = (const float*)d_in[10];  // (2,)
    float* out = (float*)d_out;                // (E,2)

    char* ws = (char*)d_ws;
    size_t o = 0;
    auto alloc = [&](size_t bytes) { char* p = ws + o; o += (bytes + 511) & ~size_t(511); return p; };

    float*  dvi    = (float*)  alloc((size_t)HV * 4);
    float*  invdvi = (float*)  alloc((size_t)HV * 4);
    float*  dei    = (float*)  alloc((size_t)HE * 4);
    __bf16* X_bf   = (__bf16*) alloc((size_t)HV * DIN * 2);
    __bf16* Wp_bf  = (__bf16*) alloc((size_t)DH * DIN * 2);
    __bf16* W0_bf  = (__bf16*) alloc((size_t)DH * DH * 2);
    __bf16* W1_bf  = (__bf16*) alloc((size_t)DH * DH * 2);
    __bf16* Hb     = (__bf16*) alloc((size_t)HV * HE * 2);   // 64 MB
    __bf16* HTb    = (__bf16*) alloc((size_t)HE * HV * 2);   // 64 MB
    float*  Xh     = (float*)  alloc((size_t)HV * DH * 4);
    __bf16* XhT_bf = (__bf16*) alloc((size_t)DH * HV * 2);
    float*  S      = (float*)  alloc((size_t)HE * DH * 4);
    __bf16* ST_bf  = (__bf16*) alloc((size_t)DH * HE * 2);
    __bf16* Y_bf   = (__bf16*) alloc((size_t)HV * DH * 2);
    float*  T      = (float*)  alloc((size_t)HV * DH * 4);
    __bf16* XsT_bf = (__bf16*) alloc((size_t)DH * HV * 2);
    float*  zE     = (float*)  alloc((size_t)HE * DH * 4);
    (void)ws_size; (void)in_sizes; (void)n_in; (void)out_size;

    // 1) degrees -> normalization vectors
    degv_kernel<<<HV, 256, 0, stream>>>(H, dvi, invdvi, HE);
    dege_kernel<<<HE / 256, 256, 0, stream>>>(H, dei, HV, HE);

    // 2) bf16 converts (NT-GEMM needs W, Wp as-is: Bt = (W^T)^T = W)
    convert_bf16_kernel<<<((size_t)HV * DIN) / 256, 256, 0, stream>>>(X, X_bf, (size_t)HV * DIN);
    convert_bf16_kernel<<<((size_t)DH * DIN) / 256, 256, 0, stream>>>(Wp, Wp_bf, (size_t)DH * DIN);
    convert_bf16_kernel<<<((size_t)DH * DH) / 256, 256, 0, stream>>>(W0, W0_bf, (size_t)DH * DH);
    convert_bf16_kernel<<<((size_t)DH * DH) / 256, 256, 0, stream>>>(W1, W1_bf, (size_t)DH * DH);

    // 3) normalized incidence in bf16, both orientations
    build_H_kernel<<<dim3(HE / 32, HV / 32), 256, 0, stream>>>(H, dvi, dei, Hb, HTb, HV, HE);

    // 4) Xh = X @ Wp^T          (A = X, Bt = Wp)
    gemm_bf16_wmma<<<dim3(DH / BN, HV / BM), 128, 0, stream>>>(
        X_bf, Wp_bf, Xh, nullptr, nullptr, HV, DH, DIN, 0);

    // 5) two layers: HHt@Xh factored as H~ @ (H~^T @ Xh)
    const __bf16* Wl[2] = { W0_bf, W1_bf };
    const float*  gg[2] = { g0, g1 };
    const float*  bb[2] = { b0, b1 };
    for (int l = 0; l < 2; ++l) {
        transpose_f32_to_bf16<<<dim3(DH / 32, HV / 32), 256, 0, stream>>>(
            Xh, XhT_bf, nullptr, HV, DH);                                 // Xh^T
        gemm_bf16_wmma<<<dim3(DH / BN, HE / BM), 128, 0, stream>>>(
            HTb, XhT_bf, S, nullptr, nullptr, HE, DH, HV, 0);             // S = H~^T Xh
        transpose_f32_to_bf16<<<dim3(DH / 32, HE / 32), 256, 0, stream>>>(
            S, ST_bf, nullptr, HE, DH);                                   // S^T
        gemm_bf16_wmma<<<dim3(DH / BN, HV / BM), 128, 0, stream>>>(
            Hb, ST_bf, nullptr, Y_bf, nullptr, HV, DH, HE, 0);            // Y = H~ S
        gemm_bf16_wmma<<<dim3(DH / BN, HV / BM), 128, 0, stream>>>(
            Y_bf, Wl[l], T, nullptr, nullptr, HV, DH, DH, 1);             // T = relu(Y W^T)
        lnres_kernel<<<HV, 256, 0, stream>>>(Xh, T, gg[l], bb[l]);        // Xh = LN(Xh+T)
    }

    // 6) pooling: z_E = dei * (H~^T (Xh / dvi)) == (H^T Xh) / d_e
    transpose_f32_to_bf16<<<dim3(DH / 32, HV / 32), 256, 0, stream>>>(
        Xh, XsT_bf, invdvi, HV, DH);                                      // (Xh/dvi)^T
    gemm_bf16_wmma<<<dim3(DH / BN, HE / BM), 128, 0, stream>>>(
        HTb, XsT_bf, zE, nullptr, dei, HE, DH, HV, 0);

    // 7) classifier + softmax
    cls_kernel<<<HE, 256, 0, stream>>>(zE, Wc, bc, out);
}
